// NetGIN_62148176773431
// MI455X (gfx1250) — matrix operand order repaired and checked by reading, not compiled, not run on hardware
//
#include <hip/hip_runtime.h>
#include <hip/hip_bf16.h>

#define N_NODES   100000
#define N_EDGES   640000
#define NUM_GRAPHS 500
#define DIM       128
#define BN_EPS    1e-5f

typedef __attribute__((ext_vector_type(16))) __bf16          bf16x16;
typedef __attribute__((ext_vector_type(8)))  float           v8f;
typedef __attribute__((ext_vector_type(16))) unsigned short  ushort16;

__device__ __forceinline__ unsigned short f2bf(float f) {
    unsigned int u = __float_as_uint(f);
    unsigned int r = (u + 0x7FFFu + ((u >> 16) & 1u)) >> 16;   // round-to-nearest-even
    return (unsigned short)r;
}

// ---------------------------------------------------------------- utilities
__global__ void k_zero(float* __restrict__ p, long n) {
    long i = (long)blockIdx.x * blockDim.x + threadIdx.x;
    if (i < n) p[i] = 0.0f;
}

// layer-1 embedding gather: t[i,:] = W1[x[i],:]
__global__ void k_gather_emb(const int* __restrict__ x, const float* __restrict__ W1,
                             float* __restrict__ t) {
    long idx = (long)blockIdx.x * blockDim.x + threadIdx.x;   // N*128 threads
    if (idx >= (long)N_NODES * DIM) return;
    int i = (int)(idx >> 7);
    int c = (int)(idx & 127);
    t[idx] = W1[(size_t)x[i] * DIM + c];
}

// scatter-add: agg[dst] += src[srcIdx], 4 channels per thread
__global__ void k_scatter_add(const float* __restrict__ src, const int* __restrict__ ei,
                              float* __restrict__ agg) {
    long tid = (long)blockIdx.x * blockDim.x + threadIdx.x;   // E*32 threads
    if (tid >= (long)N_EDGES * 32) return;
    int e  = (int)(tid >> 5);
    int c4 = (int)((tid & 31) << 2);
    int s = ei[e];
    int d = ei[N_EDGES + e];
    const float* sp = src + (size_t)s * DIM + c4;
    float*       dp = agg + (size_t)d * DIM + c4;
    atomicAdd(dp + 0, sp[0]);
    atomicAdd(dp + 1, sp[1]);
    atomicAdd(dp + 2, sp[2]);
    atomicAdd(dp + 3, sp[3]);
}

// t = relu((1+eps)*t + agg + b1)   (layer-1 pre-MLP combine)
__global__ void k_axpb_relu(float* __restrict__ t, const float* __restrict__ agg,
                            const float* __restrict__ b1, const float* __restrict__ epsPtr) {
    long idx = (long)blockIdx.x * blockDim.x + threadIdx.x;
    if (idx >= (long)N_NODES * DIM) return;
    int c = (int)(idx & 127);
    float a = 1.0f + *epsPtr;
    float v = a * t[idx] + agg[idx] + b1[c];
    t[idx] = fmaxf(v, 0.0f);
}

// repack f32 weight [K,128] (row-major) into WMMA-B bf16 layout, K-chunk major:
// Wp[kc][tn][lane][e]  with  n = tn*16 + lane%16,  k = kc*32 + (lane/16)*8 + (e%8) + (e/8)*16
// (inner tn loop in the GEMM then walks contiguous 1KB blocks)
__global__ void k_pack_w(const float* __restrict__ W, unsigned short* __restrict__ Wp, int K) {
    int idx = blockIdx.x * blockDim.x + threadIdx.x;          // K*128 threads
    if (idx >= K * DIM) return;
    int e    = idx & 15;
    int lane = (idx >> 4) & 31;
    int rest = idx >> 9;
    int tn   = rest & 7;                                      // 8 N-tiles
    int kc   = rest >> 3;
    int n = (tn << 4) + (lane & 15);
    int k = (kc << 5) + ((lane >> 4) << 3) + (e & 7) + ((e >> 3) << 4);
    Wp[idx] = f2bf(W[(size_t)k * DIM + n]);
}

// C[:, outOffset:outOffset+128] = (relu?)( bf16GEMM(Aeff, ldsW) + bias )
// Aeff = alpha*A1 + A2 (A2 optional); alpha = 1+*epsPtr if epsPtr else 1
// Packed B (K*128 bf16) is staged once per block into LDS; one wave computes a
// full 16x128 output strip: A loaded/converted once per K-chunk, B tiles pulled
// from LDS into a register array, then 8 back-to-back WMMAs.
__global__ void k_gemm_bf16(const float* __restrict__ A1, const float* __restrict__ A2,
                            const float* __restrict__ epsPtr, int lda,
                            const unsigned short* __restrict__ Wp,
                            const float* __restrict__ bias,
                            float* __restrict__ C, int outStride, int outOffset,
                            int M, int K, int doRelu) {
    extern __shared__ unsigned short ldsW[];                  // K*128 bf16

    // cooperative stage of packed weights into LDS (all threads participate)
    {
        const uint4* src = (const uint4*)Wp;
        uint4*       dst = (uint4*)ldsW;
        const int n16 = (K * DIM) >> 3;                       // 16B chunks
        for (int i = threadIdx.x; i < n16; i += blockDim.x) dst[i] = src[i];
    }
    __syncthreads();

    const int lane = threadIdx.x & 31;
    const int wave = threadIdx.x >> 5;
    const int wpb  = blockDim.x >> 5;
    const int tm   = blockIdx.x * wpb + wave;      // one M-tile per wave
    const int mT   = M >> 4;
    if (tm >= mT) return;                          // wave-uniform exit (after barrier)
    const int row = (tm << 4) + (lane & 15);
    const int hi  = lane >> 4;                     // 0/1 half-lane group
    const float alpha = epsPtr ? (1.0f + *epsPtr) : 1.0f;
    const int kcN = K >> 5;

    v8f acc[8];
#pragma unroll
    for (int t = 0; t < 8; ++t) acc[t] = v8f{0.f,0.f,0.f,0.f,0.f,0.f,0.f,0.f};

    // A operand base: lanes 0-15 hold K {0..7,16..23}; lanes 16-31 hold K {8..15,24..31}
    const float* apBase = A1 + (size_t)row * lda + hi * 8;
    const float* qpBase = A2 ? (A2 + (size_t)row * lda + hi * 8) : nullptr;

    for (int kc = 0; kc < kcN; ++kc) {
        const float* ap = apBase + (kc << 5);
        if (kc + 1 < kcN)
            __builtin_prefetch(apBase + ((kc + 1) << 5), 0, 3);   // global_prefetch

        float4 a0 = *(const float4*)(ap);
        float4 a1 = *(const float4*)(ap + 4);
        float4 a2 = *(const float4*)(ap + 16);
        float4 a3 = *(const float4*)(ap + 20);
        if (qpBase) {
            const float* qp = qpBase + (kc << 5);
            float4 q0 = *(const float4*)(qp);
            float4 q1 = *(const float4*)(qp + 4);
            float4 q2 = *(const float4*)(qp + 16);
            float4 q3 = *(const float4*)(qp + 20);
            a0.x = alpha*a0.x + q0.x; a0.y = alpha*a0.y + q0.y; a0.z = alpha*a0.z + q0.z; a0.w = alpha*a0.w + q0.w;
            a1.x = alpha*a1.x + q1.x; a1.y = alpha*a1.y + q1.y; a1.z = alpha*a1.z + q1.z; a1.w = alpha*a1.w + q1.w;
            a2.x = alpha*a2.x + q2.x; a2.y = alpha*a2.y + q2.y; a2.z = alpha*a2.z + q2.z; a2.w = alpha*a2.w + q2.w;
            a3.x = alpha*a3.x + q3.x; a3.y = alpha*a3.y + q3.y; a3.z = alpha*a3.z + q3.z; a3.w = alpha*a3.w + q3.w;
        }
        ushort16 au;
        au[0]=f2bf(a0.x); au[1]=f2bf(a0.y); au[2]=f2bf(a0.z); au[3]=f2bf(a0.w);
        au[4]=f2bf(a1.x); au[5]=f2bf(a1.y); au[6]=f2bf(a1.z); au[7]=f2bf(a1.w);
        au[8]=f2bf(a2.x); au[9]=f2bf(a2.y); au[10]=f2bf(a2.z); au[11]=f2bf(a2.w);
        au[12]=f2bf(a3.x); au[13]=f2bf(a3.y); au[14]=f2bf(a3.z); au[15]=f2bf(a3.w);
        bf16x16 a = __builtin_bit_cast(bf16x16, au);

        // pull all 8 B tiles for this K-chunk from LDS first (clauseable
        // ds_load_b128, waits stagger), then issue 8 back-to-back WMMAs
        const ushort16* bp = (const ushort16*)ldsW + ((size_t)kc << 8) + lane;
        bf16x16 bt[8];
#pragma unroll
        for (int tn = 0; tn < 8; ++tn)
            bt[tn] = __builtin_bit_cast(bf16x16, bp[tn * 32]);
#pragma unroll
        for (int tn = 0; tn < 8; ++tn)
            acc[tn] = __builtin_amdgcn_wmma_f32_16x16x32_bf16(false, a, false, bt[tn],
                                                              (short)0, acc[tn],
                                                              false, false);
    }

    // epilogue: 16x128 strip
#pragma unroll
    for (int tn = 0; tn < 8; ++tn) {
        const int c128 = (tn << 4) + (lane & 15);
        const int col  = outOffset + c128;
        float bv = bias ? bias[c128] : 0.0f;
#pragma unroll
        for (int r = 0; r < 8; ++r) {
            int orow = (tm << 4) + r + hi * 8;     // C layout: VGPR r -> M = r (+8 for hi lanes)
            float v = acc[tn][r] + bv;
            if (doRelu) v = fmaxf(v, 0.0f);
            C[(size_t)orow * outStride + col] = v;
        }
    }
}

// BN stats: per-channel sum / sumsq
__global__ void k_bn_stats(const float* __restrict__ h, float* __restrict__ stats) {
    int c = threadIdx.x;                           // 128
    float s = 0.f, ss = 0.f;
    for (int i = blockIdx.x; i < N_NODES; i += gridDim.x) {
        float v = h[(size_t)i * DIM + c];
        s += v; ss += v * v;
    }
    atomicAdd(&stats[c], s);
    atomicAdd(&stats[DIM + c], ss);
}

// BN apply; also append result into z[:, layerIdx*128:]
__global__ void k_bn_apply(float* __restrict__ h, const float* __restrict__ stats,
                           const float* __restrict__ g, const float* __restrict__ b,
                           float* __restrict__ z, int layerIdx) {
    long idx = (long)blockIdx.x * blockDim.x + threadIdx.x;
    if (idx >= (long)N_NODES * DIM) return;
    int c = (int)(idx & 127);
    int i = (int)(idx >> 7);
    const float inv = 1.0f / (float)N_NODES;
    float mu  = stats[c] * inv;
    float var = stats[DIM + c] * inv - mu * mu;
    float v = (h[idx] - mu) * rsqrtf(var + BN_EPS) * g[c] + b[c];
    h[idx] = v;
    z[(size_t)i * (4 * DIM) + layerIdx * DIM + c] = v;
}

// global mean-pool accumulation
__global__ void k_pool(const float* __restrict__ z, const int* __restrict__ batch,
                       float* __restrict__ sums, float* __restrict__ counts) {
    long idx = (long)blockIdx.x * blockDim.x + threadIdx.x;   // N*512 threads
    if (idx >= (long)N_NODES * 4 * DIM) return;
    int i = (int)(idx >> 9);
    int c = (int)(idx & 511);
    int gi = batch[i];
    atomicAdd(&sums[(size_t)gi * 512 + c], z[idx]);
    if (c == 0) atomicAdd(&counts[gi], 1.0f);
}

// FC head: 1 block (128 threads) per graph
__global__ void k_head(const float* __restrict__ sums, const float* __restrict__ counts,
                       const float* __restrict__ W1, const float* __restrict__ b1,
                       const float* __restrict__ W2, const float* __restrict__ b2,
                       const float* __restrict__ W3, const float* __restrict__ b3,
                       const float* __restrict__ W4, const float* __restrict__ b4,
                       float* __restrict__ out) {
    __shared__ float g[512];
    __shared__ float a[128];
    __shared__ float t[128];
    __shared__ float red[128];
    int gi = blockIdx.x, tid = threadIdx.x;
    float cnt = fmaxf(counts[gi], 1.0f);
    for (int k = tid; k < 512; k += 128) g[k] = sums[(size_t)gi * 512 + k] / cnt;
    __syncthreads();
    float acc = b1[tid];
    for (int k = 0; k < 512; ++k) acc += g[k] * W1[k * 128 + tid];
    a[tid] = fmaxf(acc, 0.0f); __syncthreads();
    acc = b2[tid];
    for (int k = 0; k < 128; ++k) acc += a[k] * W2[k * 128 + tid];
    t[tid] = fmaxf(acc, 0.0f); __syncthreads();
    acc = b3[tid];
    for (int k = 0; k < 128; ++k) acc += t[k] * W3[k * 128 + tid];
    a[tid] = fmaxf(acc, 0.0f); __syncthreads();
    red[tid] = a[tid] * W4[tid]; __syncthreads();
    for (int s = 64; s > 0; s >>= 1) { if (tid < s) red[tid] += red[tid + s]; __syncthreads(); }
    if (tid == 0) out[gi] = red[0] + b4[0];
}

// ---------------------------------------------------------------- host side
static inline int cdiv(long a, int b) { return (int)((a + b - 1) / b); }

extern "C" void kernel_launch(void* const* d_in, const int* in_sizes, int n_in,
                              void* d_out, int out_size, void* d_ws, size_t ws_size,
                              hipStream_t stream) {
    (void)in_sizes; (void)n_in; (void)out_size; (void)ws_size;

    const int* x     = (const int*)d_in[0];
    // setup_inputs order: e1l, e2l, e1g, e2g
    const int* eBuf[4] = { (const int*)d_in[1], (const int*)d_in[2],
                           (const int*)d_in[3], (const int*)d_in[4] };
    const int* batch = (const int*)d_in[5];
    auto F = [&](int i) { return (const float*)d_in[i]; };

    // params flattened in JAX sorted-key order, leaves at d_in[6..]
    // bn{L}: base 6+(L-1)*2 -> {b, g}
    // conv{L}_{1_g,1_l,2_g,2_l}: base 14+(L-1)*20 + {0,5,10,15} -> {W1, W2, b1, b2, eps}
    // fc1..fc4: 94..101 -> {W, b} each
    // mlp{L}: base 102+(L-1)*4 -> {W1, W2, b1, b2}
    const int BN0 = 6, CV0 = 14, FC0 = 94, ML0 = 102;

    // cat order [x1, x3, x2, x4] = branches (1_l, 1_g, 2_l, 2_g)
    const int slotEdge[4] = { 0, 2, 1, 3 };        // index into eBuf
    const int slotConv[4] = { 5, 0, 15, 10 };      // offset within sorted conv block

    // ---- workspace carve-up (all 64B-aligned)
    float* h      = (float*)d_ws;                                   // N*128
    float* agg    = h      + (size_t)N_NODES * DIM;                 // N*128
    float* tmp    = agg    + (size_t)N_NODES * DIM;                 // N*128
    float* cat    = tmp    + (size_t)N_NODES * DIM;                 // N*512
    float* z      = cat    + (size_t)N_NODES * 4 * DIM;             // N*512
    float* stats  = z      + (size_t)N_NODES * 4 * DIM;             // 256
    float* sums   = stats  + 256;                                   // 500*512
    float* counts = sums   + (size_t)NUM_GRAPHS * 512;              // 512 (incl. pad)
    unsigned short* wp = (unsigned short*)(counts + 512);           // up to 512*128

    const long ND  = (long)N_NODES * DIM;
    const long ND4 = (long)N_NODES * 4 * DIM;
    const long ESC = (long)N_EDGES * 32;
    const int mT   = N_NODES / 16;                                  // 6250 M-tiles
    const int gemmBlocks = cdiv(mT, 4);                             // 4 waves/block
    const size_t lds128 = (size_t)DIM * DIM * sizeof(unsigned short);      // 32 KB
    const size_t lds512 = (size_t)4 * DIM * DIM * sizeof(unsigned short);  // 128 KB

    for (int L = 1; L <= 4; ++L) {
        const int cvBase = CV0 + (L - 1) * 20;
        for (int slot = 0; slot < 4; ++slot) {
            const int p = cvBase + slotConv[slot];
            const float *W1 = F(p), *W2 = F(p + 1), *b1 = F(p + 2),
                        *b2 = F(p + 3), *eps = F(p + 4);
            const int* ei = eBuf[slotEdge[slot]];

            k_zero<<<cdiv(ND, 256), 256, 0, stream>>>(agg, ND);
            if (L == 1) {
                k_gather_emb<<<cdiv(ND, 256), 256, 0, stream>>>(x, W1, tmp);
                k_scatter_add<<<cdiv(ESC, 256), 256, 0, stream>>>(tmp, ei, agg);
                k_axpb_relu<<<cdiv(ND, 256), 256, 0, stream>>>(tmp, agg, b1, eps);
            } else {
                k_scatter_add<<<cdiv(ESC, 256), 256, 0, stream>>>(h, ei, agg);
                // GEMM1: relu(((1+eps)h + agg) @ W1 + b1) -> tmp
                k_pack_w<<<cdiv(DIM * DIM, 256), 256, 0, stream>>>(W1, wp, DIM);
                k_gemm_bf16<<<gemmBlocks, 128, lds128, stream>>>(
                    h, agg, eps, DIM, wp, b1, tmp, DIM, 0, N_NODES, DIM, 1);
            }
            // GEMM2: relu(tmp @ W2 + b2) -> cat[:, slot*128:]
            k_pack_w<<<cdiv(DIM * DIM, 256), 256, 0, stream>>>(W2, wp, DIM);
            k_gemm_bf16<<<gemmBlocks, 128, lds128, stream>>>(
                tmp, nullptr, nullptr, DIM, wp, b2, cat, 4 * DIM, slot * DIM,
                N_NODES, DIM, 1);
        }
        // fusion MLP: relu(cat @ mW1 + mb1) @ mW2 + mb2 -> h (pre-BN)
        const int m = ML0 + (L - 1) * 4;
        k_pack_w<<<cdiv(4 * DIM * DIM, 256), 256, 0, stream>>>(F(m), wp, 4 * DIM);
        k_gemm_bf16<<<gemmBlocks, 128, lds512, stream>>>(
            cat, nullptr, nullptr, 4 * DIM, wp, F(m + 2), tmp, DIM, 0,
            N_NODES, 4 * DIM, 1);
        k_pack_w<<<cdiv(DIM * DIM, 256), 256, 0, stream>>>(F(m + 1), wp, DIM);
        k_gemm_bf16<<<gemmBlocks, 128, lds128, stream>>>(
            tmp, nullptr, nullptr, DIM, wp, F(m + 3), h, DIM, 0,
            N_NODES, DIM, 0);
        // batchnorm (+ append to z)
        k_zero<<<1, 256, 0, stream>>>(stats, 256);
        k_bn_stats<<<512, 128, 0, stream>>>(h, stats);
        const int bnB = BN0 + (L - 1) * 2;
        k_bn_apply<<<cdiv(ND, 256), 256, 0, stream>>>(h, stats, F(bnB + 1), F(bnB), z, L - 1);
    }

    // global mean pool + head
    k_zero<<<cdiv((long)NUM_GRAPHS * 512 + 512, 256), 256, 0, stream>>>(
        sums, (long)NUM_GRAPHS * 512 + 512);
    k_pool<<<cdiv(ND4, 256), 256, 0, stream>>>(z, batch, sums, counts);
    k_head<<<NUM_GRAPHS, 128, 0, stream>>>(
        sums, counts,
        F(FC0), F(FC0 + 1), F(FC0 + 2), F(FC0 + 3),
        F(FC0 + 4), F(FC0 + 5), F(FC0 + 6), F(FC0 + 7),
        (float*)d_out);
}
